// ViTDeformAttention_29283087024800
// MI455X (gfx1250) — matrix operand order repaired and checked by reading, not compiled, not run on hardware
//
#include <hip/hip_runtime.h>
#include <hip/hip_fp16.h>

// ---------------------------------------------------------------------------
// Types / WMMA helpers (CDNA5 gfx1250, wave32)
// ---------------------------------------------------------------------------
typedef __attribute__((ext_vector_type(16))) _Float16 v16h;
typedef __attribute__((ext_vector_type(8)))  _Float16 v8h;
typedef __attribute__((ext_vector_type(8)))  float    v8f;

__device__ __forceinline__ v16h ld_frag(const _Float16* p0, const _Float16* p1) {
  v8h lo = *reinterpret_cast<const v8h*>(p0);
  v8h hi = *reinterpret_cast<const v8h*>(p1);
  return __builtin_shufflevector(lo, hi, 0,1,2,3,4,5,6,7,8,9,10,11,12,13,14,15);
}
__device__ __forceinline__ v16h ld_frag16(const _Float16* p) { return ld_frag(p, p + 8); }

__device__ __forceinline__ v8f wmma_f16(v16h a, v16h b, v8f c) {
  // D = A(16x32 f16) * B(32x16 f16) + C(16x16 f32)
  return __builtin_amdgcn_wmma_f32_16x16x32_f16(false, a, false, b, (short)0, c,
                                                false, false);
}

__device__ __forceinline__ float wred_add32(float v) {
#pragma unroll
  for (int m = 1; m <= 16; m <<= 1) v += __shfl_xor(v, m, 32);
  return v;
}

// ---------------------------------------------------------------------------
// Problem constants
// ---------------------------------------------------------------------------
#define NTOK   197          // h*w+1
#define BT     64           // b*t
#define DIMC   384
#define NQROW  (NTOK * BT)  // 12608 rows (multiple of 64)
#define NKROW  (8 * 1568)   // 12544 rows (multiple of 64)
#define NKV    1568         // t*h*w keys
#define MQ     1576         // t * NTOK queries
#define MQPAD  1584         // padded to 16
#define NBH    48           // b * heads

// ---------------------------------------------------------------------------
// f32 -> f16 elementwise convert
// ---------------------------------------------------------------------------
__global__ void cvt_f16_k(const float* __restrict__ in, _Float16* __restrict__ out, int n) {
  int i = blockIdx.x * 256 + threadIdx.x;
  if (i < n) out[i] = (_Float16)in[i];
}

// ---------------------------------------------------------------------------
// Repack q (f32 [n][bt][384]) -> f16 attention-A layout [bh][m_pad][64], *SCALE
// ---------------------------------------------------------------------------
__global__ void repack_qatt_k(const float* __restrict__ q, _Float16* __restrict__ qatt) {
  int i = blockIdx.x * 256 + threadIdx.x;
  if (i >= NBH * MQPAD * 64) return;
  int hc = i & 63;
  int m  = (i >> 6) % MQPAD;
  int bh = i / (MQPAD * 64);
  int b = bh / 6, head = bh % 6;
  float v = 0.f;
  if (m < MQ) {
    int t = m / NTOK, ni = m % NTOK;
    v = q[(size_t)(ni * BT + b * 8 + t) * DIMC + head * 64 + hc] * 0.125f; // SCALE=HC^-0.5
  }
  qatt[i] = (_Float16)v;
}

// ---------------------------------------------------------------------------
// GEMM: A[M x 384] f16 row major, W[384 x 384] f16 in [out][in] layout
// (so B[k][n] = W[n][k]).  4 waves per block share one N-tile: the 16x384 W
// tile is staged into LDS once per block with global_load_async_to_lds_b128
// (ASYNCcnt), then each wave computes a 16x16 tile reading B from LDS.
// mode 0: outf f32 [M x 384]
// mode 1: outh f16 K-attn layout  [bh][nk][64]
// mode 2: outh f16 V-attn layout  [bh][64][nk]
// ---------------------------------------------------------------------------
__global__ void __launch_bounds__(128)
gemm384_k(const _Float16* __restrict__ A, const _Float16* __restrict__ W,
          const float* __restrict__ bias, float* __restrict__ outf,
          _Float16* __restrict__ outh, int M, int mode) {
  __shared__ _Float16 lds_w[16 * DIMC];   // 12 KB: one W N-tile
  const int lane = threadIdx.x & 31;
  const int wid  = threadIdx.x >> 5;
  const int tn   = blockIdx.x % 24;                 // N tile (384/16)
  const int tm   = (blockIdx.x / 24) * 4 + wid;     // M tile, 4 per block
  // ---- async-stage W tile (16 rows x 384 halfs = 12288 B) into LDS --------
  {
    uint64_t wbase = (uint64_t)(uintptr_t)(W + (size_t)tn * 16 * DIMC);
    uint32_t lbase = (uint32_t)(uintptr_t)(&lds_w[0]);
#pragma unroll
    for (int j = 0; j < 6; ++j) {                   // 128 lanes * 6 * 16B = 12 KB
      uint32_t off16 = (uint32_t)(threadIdx.x + 128 * j) * 16u;
      asm volatile("global_load_async_to_lds_b128 %0, %1, %2"
                   :: "v"(lbase + off16), "v"(off16), "s"(wbase)
                   : "memory");
    }
    asm volatile("s_wait_asynccnt 0x0" ::: "memory");
    __syncthreads();
  }
  // ---- 16x16 tile per wave, K swept in 12 WMMA steps ----------------------
  const int l15   = lane & 15;
  const int abase = (lane < 16) ? 0 : 16;  // A: K contiguous per lane half
  const int kb    = (lane < 16) ? 0 : 16;  // B: K contiguous per lane half
  const _Float16* arow = A + (size_t)(tm * 16 + l15) * DIMC;
  const _Float16* brow = &lds_w[l15 * DIMC];
  v8f c = {0.f, 0.f, 0.f, 0.f, 0.f, 0.f, 0.f, 0.f};
#pragma unroll
  for (int k0 = 0; k0 < DIMC; k0 += 32) {
    v16h a = ld_frag(arow + k0 + abase, arow + k0 + 8 + abase);   // global b128 x2
    v16h b = ld_frag16(brow + k0 + kb);                           // ds b128 x2
    c = wmma_f16(a, b, c);
  }
  const int   gn    = tn * 16 + l15;
  const float bval  = bias[gn];
  const int   mbase = tm * 16 + ((lane < 16) ? 0 : 8);
#pragma unroll
  for (int r = 0; r < 8; ++r) {
    float val = c[r] + bval;
    int   gm  = mbase + r;
    if (mode == 0) {
      outf[(size_t)gm * DIMC + gn] = val;
    } else {
      int bb = gm / NKV, nk = gm % NKV;
      int head = gn >> 6, hc = gn & 63;
      int bh = bb * 6 + head;
      if (mode == 1) outh[((size_t)bh * NKV + nk) * 64 + hc] = (_Float16)val;  // K
      else           outh[((size_t)bh * 64 + hc) * NKV + nk] = (_Float16)val;  // V
    }
  }
}

// ---------------------------------------------------------------------------
// Offset path: depthwise conv3d 3x3x3 + LN(channels) + GELU + 1x1 conv -> 3
// offsets -> tanh*rng + ref grid -> trilinear grid-sample of x -> f16 KV input
// One wave per (bg, t, y, x) position; 2 channels per lane.
// ---------------------------------------------------------------------------
__global__ void __launch_bounds__(256)
conv_offset_sample_k(const float* __restrict__ x, const float* __restrict__ q,
                     const float* __restrict__ dww, const float* __restrict__ dwb,
                     const float* __restrict__ lnw, const float* __restrict__ lnb,
                     const float* __restrict__ pww, _Float16* __restrict__ kvin) {
  const int lane = threadIdx.x & 31;
  const int wid  = threadIdx.x >> 5;
  const int p    = blockIdx.x * 8 + wid;   // 0..1567 = (t,y,x)
  const int bg   = blockIdx.y;             // 0..47
  const int b = bg / 6, g = bg % 6;
  const int t = p / 196, rem = p % 196, y = rem / 14, xw = rem % 14;
  const int gc0 = lane * 2;

  // depthwise conv3d (zero pad) + bias
  float o0 = dwb[gc0], o1 = dwb[gc0 + 1];
#pragma unroll
  for (int dz = -1; dz <= 1; ++dz) {
    int tz = t + dz; if (tz < 0 || tz >= 8) continue;
#pragma unroll
    for (int dy = -1; dy <= 1; ++dy) {
      int yy = y + dy; if (yy < 0 || yy >= 14) continue;
#pragma unroll
      for (int dx = -1; dx <= 1; ++dx) {
        int xx = xw + dx; if (xx < 0 || xx >= 14) continue;
        int widx = (dz + 1) * 9 + (dy + 1) * 3 + (dx + 1);
        const float* src = q + (size_t)((1 + yy * 14 + xx) * BT + b * 8 + tz) * DIMC
                             + g * 64 + gc0;
        o0 += dww[gc0 * 27 + widx]       * src[0];
        o1 += dww[(gc0 + 1) * 27 + widx] * src[1];
      }
    }
  }
  // LayerNorm over 64 channels
  float mu  = wred_add32(o0 + o1) * (1.f / 64.f);
  float var = wred_add32(o0 * o0 + o1 * o1) * (1.f / 64.f) - mu * mu;
  float inv = rsqrtf(var + 1e-5f);
  float a0 = (o0 - mu) * inv * lnw[gc0]     + lnb[gc0];
  float a1 = (o1 - mu) * inv * lnw[gc0 + 1] + lnb[gc0 + 1];
  // exact GELU
  a0 = 0.5f * a0 * (1.f + erff(a0 * 0.70710678118f));
  a1 = 0.5f * a1 * (1.f + erff(a1 * 0.70710678118f));
  // pointwise 64 -> 3
  float off0 = wred_add32(pww[0 * 64 + gc0] * a0 + pww[0 * 64 + gc0 + 1] * a1);
  float off1 = wred_add32(pww[1 * 64 + gc0] * a0 + pww[1 * 64 + gc0 + 1] * a1);
  float off2 = wred_add32(pww[2 * 64 + gc0] * a0 + pww[2 * 64 + gc0 + 1] * a1);
  // pos = tanh(off)*rng + ref    (channel order z,y,x)
  float pz = tanhf(off0) * 0.25f       + ((t  + 0.5f) / 8.f ) * 2.f - 1.f;
  float py = tanhf(off1) * (1.f / 7.f) + ((y  + 0.5f) / 14.f) * 2.f - 1.f;
  float px = tanhf(off2) * (1.f / 7.f) + ((xw + 0.5f) / 14.f) * 2.f - 1.f;
  // trilinear sample, align_corners=True, zero padding
  float gx = (px + 1.f) * 0.5f * 13.f;
  float gy = (py + 1.f) * 0.5f * 13.f;
  float gz = (pz + 1.f) * 0.5f * 7.f;
  float x0f = floorf(gx), y0f = floorf(gy), z0f = floorf(gz);
  float wx1 = gx - x0f, wy1 = gy - y0f, wz1 = gz - z0f;
  float wx0 = 1.f - wx1, wy0 = 1.f - wy1, wz0 = 1.f - wz1;
  int x0 = (int)x0f, y0 = (int)y0f, z0 = (int)z0f;
  float acc0 = 0.f, acc1 = 0.f;
#pragma unroll
  for (int cz = 0; cz < 2; ++cz) {
    int zi = z0 + cz; if (zi < 0 || zi >= 8) continue;
    float wz = cz ? wz1 : wz0;
#pragma unroll
    for (int cy = 0; cy < 2; ++cy) {
      int yi = y0 + cy; if (yi < 0 || yi >= 14) continue;
      float wzy = wz * (cy ? wy1 : wy0);
#pragma unroll
      for (int cx = 0; cx < 2; ++cx) {
        int xi = x0 + cx; if (xi < 0 || xi >= 14) continue;
        float wgt = wzy * (cx ? wx1 : wx0);
        const float* src = x + (size_t)((1 + yi * 14 + xi) * BT + b * 8 + zi) * DIMC
                             + g * 64 + gc0;
        acc0 += wgt * src[0];
        acc1 += wgt * src[1];
      }
    }
  }
  _Float16* dst = kvin + ((size_t)b * NKV + p) * DIMC + g * 64 + gc0;
  dst[0] = (_Float16)acc0;
  dst[1] = (_Float16)acc1;
}

// ---------------------------------------------------------------------------
// Flash attention: 4 waves per block, one (bh, 16-row m-tile) per wave.
// Per 32-key step: 4x WMMA for S (K=64), online softmax in registers,
// P transposed through a wave-private LDS tile (wave-local s_wait_dscnt,
// no block barrier needed), 4x WMMA for O (N=64).
// ---------------------------------------------------------------------------
__global__ void __launch_bounds__(128)
flash_attn_k(const _Float16* __restrict__ qatt, const _Float16* __restrict__ katt,
             const _Float16* __restrict__ vatt, _Float16* __restrict__ oatt) {
  __shared__ _Float16 pl[4][16 * 32];      // per-wave P tile (1 KB each)
  const int lane  = threadIdx.x & 31;
  const int wid   = threadIdx.x >> 5;
  const int l15   = lane & 15;
  int mt = blockIdx.x * 4 + wid;           // 0..99 -> clamp (dup writes same data)
  const bool live = (mt < MQPAD / 16);
  if (!live) mt = MQPAD / 16 - 1;
  const int bh    = blockIdx.y;            // 0..47
  const int b     = bh / 6, head = bh % 6;
  const int abase = (lane < 16) ? 0 : 16;
  const int kb    = (lane < 16) ? 0 : 16;
  const int prow  = (lane < 16) ? 0 : 8;
  _Float16* plw   = pl[wid];

  const _Float16* qrow = qatt + ((size_t)bh * MQPAD + mt * 16 + l15) * 64;
  v16h aq0 = ld_frag(qrow + 0  + abase, qrow + 8 + abase);    // hc 0..31
  v16h aq1 = ld_frag(qrow + 32 + abase, qrow + 40 + abase);   // hc 32..63

  const v8f zf = {0.f, 0.f, 0.f, 0.f, 0.f, 0.f, 0.f, 0.f};
  v8f oacc[4] = {zf, zf, zf, zf};
  float mrow[8], lrow[8];
#pragma unroll
  for (int r = 0; r < 8; ++r) { mrow[r] = -1e30f; lrow[r] = 0.f; }

  const _Float16* kbp = katt + (size_t)bh * NKV * 64;
  const _Float16* vbp = vatt + (size_t)bh * 64 * NKV;

  for (int n0 = 0; n0 < NKV; n0 += 32) {
    const _Float16* kr0 = kbp + (size_t)(n0 + l15) * 64 + kb;
    const _Float16* kr1 = kbp + (size_t)(n0 + 16 + l15) * 64 + kb;
    v8f s0 = zf, s1 = zf;
    s0 = wmma_f16(aq0, ld_frag16(kr0), s0);
    s0 = wmma_f16(aq1, ld_frag16(kr0 + 32), s0);
    s1 = wmma_f16(aq0, ld_frag16(kr1), s1);
    s1 = wmma_f16(aq1, ld_frag16(kr1 + 32), s1);

    // online softmax (rows striped: vgpr r <-> row prow+r, 16 lanes share a row)
    float rmax[8], rs[8], alpha[8];
#pragma unroll
    for (int r = 0; r < 8; ++r) rmax[r] = fmaxf(s0[r], s1[r]);
#pragma unroll
    for (int msk = 1; msk <= 8; msk <<= 1)
#pragma unroll
      for (int r = 0; r < 8; ++r)
        rmax[r] = fmaxf(rmax[r], __shfl_xor(rmax[r], msk, 32));
#pragma unroll
    for (int r = 0; r < 8; ++r) {
      float mn = fmaxf(mrow[r], rmax[r]);
      alpha[r] = __expf(mrow[r] - mn);
      mrow[r]  = mn;
      s0[r] = __expf(s0[r] - mn);
      s1[r] = __expf(s1[r] - mn);
      rs[r] = s0[r] + s1[r];
    }
#pragma unroll
    for (int msk = 1; msk <= 8; msk <<= 1)
#pragma unroll
      for (int r = 0; r < 8; ++r) rs[r] += __shfl_xor(rs[r], msk, 32);
#pragma unroll
    for (int r = 0; r < 8; ++r) {
      lrow[r] = lrow[r] * alpha[r] + rs[r];
      oacc[0][r] *= alpha[r]; oacc[1][r] *= alpha[r];
      oacc[2][r] *= alpha[r]; oacc[3][r] *= alpha[r];
    }
    // transpose P (C layout -> A layout) through wave-private LDS tile.
    // LDS ops of one wave complete in order; s_wait_dscnt covers the
    // store->load visibility, no cross-wave barrier required.
#pragma unroll
    for (int r = 0; r < 8; ++r) {
      plw[(prow + r) * 32 + l15]      = (_Float16)s0[r];
      plw[(prow + r) * 32 + 16 + l15] = (_Float16)s1[r];
    }
    asm volatile("s_wait_dscnt 0x0" ::: "memory");
    v16h ap = ld_frag(&plw[l15 * 32 + abase], &plw[l15 * 32 + 8 + abase]);
    // O += P (16x32) * V (32x64)
#pragma unroll
    for (int ct = 0; ct < 4; ++ct) {
      const _Float16* vr = vbp + (size_t)(ct * 16 + l15) * NKV + n0 + kb;
      oacc[ct] = wmma_f16(ap, ld_frag16(vr), oacc[ct]);
    }
  }

  // epilogue: divide by l, scatter to [n][bt][c] layout (f16, A of final GEMM)
#pragma unroll
  for (int ct = 0; ct < 4; ++ct) {
    int gn = head * 64 + ct * 16 + l15;
#pragma unroll
    for (int r = 0; r < 8; ++r) {
      int m = mt * 16 + prow + r;
      if (m < MQ) {
        int t = m / NTOK, ni = m % NTOK;
        oatt[(size_t)(ni * BT + b * 8 + t) * DIMC + gn] = (_Float16)(oacc[ct][r] / lrow[r]);
      }
    }
  }
}

// ---------------------------------------------------------------------------
// Launch
// ---------------------------------------------------------------------------
extern "C" void kernel_launch(void* const* d_in, const int* in_sizes, int n_in,
                              void* d_out, int out_size, void* d_ws, size_t ws_size,
                              hipStream_t stream) {
  const float* x   = (const float*)d_in[0];
  const float* wq  = (const float*)d_in[1];
  const float* bq  = (const float*)d_in[2];
  const float* wk  = (const float*)d_in[3];
  const float* bk  = (const float*)d_in[4];
  const float* wv  = (const float*)d_in[5];
  const float* bv  = (const float*)d_in[6];
  const float* wo  = (const float*)d_in[7];
  const float* bo  = (const float*)d_in[8];
  const float* dww = (const float*)d_in[9];
  const float* dwb = (const float*)d_in[10];
  const float* lnw = (const float*)d_in[11];
  const float* lnb = (const float*)d_in[12];
  const float* pww = (const float*)d_in[13];
  float* out = (float*)d_out;

  char*  ws  = (char*)d_ws;
  size_t off = 0;
  auto carve = [&](size_t bytes) -> char* {
    char* p = ws + off;
    off += (bytes + 255) & ~(size_t)255;
    return p;
  };
  _Float16* h_x    = (_Float16*)carve((size_t)NQROW * DIMC * 2);
  _Float16* h_wq   = (_Float16*)carve((size_t)DIMC * DIMC * 2);
  _Float16* h_wk   = (_Float16*)carve((size_t)DIMC * DIMC * 2);
  _Float16* h_wv   = (_Float16*)carve((size_t)DIMC * DIMC * 2);
  _Float16* h_wo   = (_Float16*)carve((size_t)DIMC * DIMC * 2);
  float*    qf     = (float*)   carve((size_t)NQROW * DIMC * 4);
  _Float16* h_qatt = (_Float16*)carve((size_t)NBH * MQPAD * 64 * 2);
  _Float16* h_kv   = (_Float16*)carve((size_t)NKROW * DIMC * 2);
  _Float16* katt   = (_Float16*)carve((size_t)NBH * NKV * 64 * 2);
  _Float16* vatt   = (_Float16*)carve((size_t)NBH * NKV * 64 * 2);
  _Float16* h_oatt = (_Float16*)carve((size_t)NQROW * DIMC * 2);
  (void)in_sizes; (void)n_in; (void)out_size; (void)ws_size;

  const int nx = NQROW * DIMC;     // 4,841,472
  const int nw = DIMC * DIMC;      // 147,456
  cvt_f16_k<<<(nx + 255) / 256, 256, 0, stream>>>(x,  h_x,  nx);
  cvt_f16_k<<<(nw + 255) / 256, 256, 0, stream>>>(wq, h_wq, nw);
  cvt_f16_k<<<(nw + 255) / 256, 256, 0, stream>>>(wk, h_wk, nw);
  cvt_f16_k<<<(nw + 255) / 256, 256, 0, stream>>>(wv, h_wv, nw);
  cvt_f16_k<<<(nw + 255) / 256, 256, 0, stream>>>(wo, h_wo, nw);

  // q = x @ wq^T + bq  (f32, feeds conv path and attention repack)
  gemm384_k<<<(NQROW / 64) * 24, 128, 0, stream>>>(h_x, h_wq, bq, qf, nullptr,
                                                   NQROW, 0);
  const int nq = NBH * MQPAD * 64;
  repack_qatt_k<<<(nq + 255) / 256, 256, 0, stream>>>(qf, h_qatt);

  // offsets + grid sample -> f16 KV GEMM input [b][nk][384]
  conv_offset_sample_k<<<dim3(196, 48), 256, 0, stream>>>(x, qf, dww, dwb, lnw, lnb,
                                                          pww, h_kv);

  // k/v projections into attention layouts
  gemm384_k<<<(NKROW / 64) * 24, 128, 0, stream>>>(h_kv, h_wk, bk, nullptr, katt,
                                                   NKROW, 1);
  gemm384_k<<<(NKROW / 64) * 24, 128, 0, stream>>>(h_kv, h_wv, bv, nullptr, vatt,
                                                   NKROW, 2);

  // fused attention: 4 m-tiles per block
  flash_attn_k<<<dim3((MQPAD / 16 + 3) / 4, NBH), 128, 0, stream>>>(h_qatt, katt,
                                                                    vatt, h_oatt);

  // final projection straight into d_out ([n][bt][c] f32)
  gemm384_k<<<(NQROW / 64) * 24, 128, 0, stream>>>(h_oatt, h_wo, bo, out, nullptr,
                                                   NQROW, 0);
}
